// GraphEmbedding_8641474199754
// MI455X (gfx1250) — matrix operand order repaired and checked by reading, not compiled, not run on hardware
//
#include <hip/hip_runtime.h>
#include <hip/hip_bf16.h>

// ---------------------------------------------------------------------------
// MI455X (gfx1250) implementation. wave32, WMMA 16x16x32 bf16 (f32 accum).
// Layouts used by this code:
//   A (16x32, 16-bit): lane l holds M=l&15; elem e -> K=16*(l>>4)+e
//                      (two 8-elem loads of one contiguous 16-elem run)
//   B (32x16, 16-bit): lane l holds N=l&15; elem e -> K=16*(l>>4)+e
//                      (one contiguous 16-elem load from the "row" of W)
//   C/D (16x16, f32) : vgpr r, lane l -> M=r+8*(l>>4), N=l&15
// ---------------------------------------------------------------------------

typedef __bf16 bf16;
typedef __attribute__((ext_vector_type(16))) __bf16 v16bf;
typedef __attribute__((ext_vector_type(8)))  __bf16 v8bf;
typedef __attribute__((ext_vector_type(8)))  float  v8f;

#define WMMA_BF16(a, b, c) \
  __builtin_amdgcn_wmma_f32_16x16x32_bf16(false, (a), false, (b), (short)0, (c), false, false)

__device__ __forceinline__ v16bf cat8(v8bf lo, v8bf hi) {
  v16bf r;
#pragma unroll
  for (int i = 0; i < 8; ++i) { r[i] = lo[i]; r[i + 8] = hi[i]; }
  return r;
}

__device__ __forceinline__ v8f v8f_zero() {
  v8f z = {0.f, 0.f, 0.f, 0.f, 0.f, 0.f, 0.f, 0.f};
  return z;
}

// ---------------------------------------------------------------------------
// Generic tiled GEMM:  Out[M,Nt] = A[M,K] @ W[Nt,K]^T + bias (+addsrc)
// A, W in bf16; accumulate f32 in WMMA; optional f32 and/or bf16 outputs.
// One wave per 16x16 output tile; block = 4 waves spanning 64 output cols.
// All operands are L2-resident (<1MB weights, <=6MB activations), so direct
// global b128 loads feeding WMMA is the right shape; no LDS staging needed.
// ---------------------------------------------------------------------------
__global__ void gemm_bf16_kernel(const bf16* __restrict__ A, const bf16* __restrict__ W,
                                 const float* __restrict__ bias,
                                 float* __restrict__ outF, bf16* __restrict__ outB,
                                 const float* __restrict__ addsrc,
                                 int K, int Nt) {
  const int lane = threadIdx.x & 31;
  const int wv   = threadIdx.x >> 5;
  const int half = lane >> 4;
  const int lm   = lane & 15;
  const int tm   = blockIdx.x;
  const int tn   = blockIdx.y * 4 + wv;

  const bf16* arow = A + (size_t)(tm * 16 + lm) * K;   // A-layout: this lane's M row
  const bf16* wrow = W + (size_t)(tn * 16 + lm) * K;   // B-layout: this lane's N row of W

  v8f acc = v8f_zero();
  for (int k0 = 0; k0 < K; k0 += 32) {
    v16bf a = cat8(*(const v8bf*)(arow + k0 + 16 * half),
                   *(const v8bf*)(arow + k0 + 16 * half + 8));
    v16bf b = cat8(*(const v8bf*)(wrow + k0 + 16 * half),
                   *(const v8bf*)(wrow + k0 + 16 * half + 8));
    acc = WMMA_BF16(a, b, acc);
  }

  const int   ncol = tn * 16 + lm;
  const float bv   = bias ? bias[ncol] : 0.f;
#pragma unroll
  for (int r = 0; r < 8; ++r) {
    const size_t mrow = (size_t)tm * 16 + r + 8 * half;
    float v = acc[r] + bv;
    if (addsrc) v += addsrc[mrow * Nt + ncol];
    if (outF) outF[mrow * Nt + ncol] = v;
    if (outB) outB[mrow * Nt + ncol] = (bf16)v;
  }
}

// ---------------------------------------------------------------------------
// dim_reduce: per node n, T[o,p] = conv_w[o,:] . feats[n,:,p]  (o=256, p=16)
// One 512-thread block (16 waves) per node; wave wv owns o-tile wv.
// Per 32-channel K-step the block does ONE coalesced 2KB f32 load of the
// feats chunk, converts to bf16, stores it transposed [p][c] in LDS so each
// wave's B fragment is two contiguous ds_load_b128. This reads each node's
// 16KB feature block once per node (64MB total) instead of 16x with
// stride-64B scalar loads (~1GB).
// Epilogue: +conv_b, BN(eval) affine, ReLU, spatial mean via shfl butterfly.
// ---------------------------------------------------------------------------
__global__ void dimreduce_kernel(const float* __restrict__ feats,
                                 const bf16* __restrict__ convw,
                                 const float* __restrict__ convb,
                                 const float* __restrict__ bng, const float* __restrict__ bnb,
                                 const float* __restrict__ bnm, const float* __restrict__ bnv,
                                 bf16* __restrict__ x_bf) {
  __shared__ __attribute__((aligned(16))) bf16 fbufT[16 * 32];   // [p][c-chunk]

  const int tid  = threadIdx.x;       // 0..511
  const int wv   = tid >> 5;          // o-tile 0..15
  const int lane = tid & 31;
  const int half = lane >> 4;
  const int lm   = lane & 15;
  const int n    = blockIdx.x;

  const float* fb   = feats + (size_t)n * 4096;                  // [c=256][p=16]
  const bf16*  arow = convw + (size_t)(wv * 16 + lm) * 256;

  const int cc = tid >> 4;            // 0..31 : channel within chunk
  const int pp = tid & 15;            // 0..15 : spatial position

  v8f acc = v8f_zero();
  for (int k0 = 0; k0 < 256; k0 += 32) {
    // coalesced cooperative load: 512 consecutive f32, transpose to [p][c] bf16
    fbufT[pp * 32 + cc] = (bf16)fb[(k0 + cc) * 16 + pp];
    __syncthreads();

    v16bf a = cat8(*(const v8bf*)(arow + k0 + 16 * half),
                   *(const v8bf*)(arow + k0 + 16 * half + 8));
    const bf16* br = fbufT + lm * 32 + 16 * half;                // B[k=c][n=p=lm]
    v16bf b = cat8(*(const v8bf*)br, *(const v8bf*)(br + 8));
    acc = WMMA_BF16(a, b, acc);
    __syncthreads();
  }

#pragma unroll
  for (int r = 0; r < 8; ++r) {
    const int   o = wv * 16 + r + 8 * half;
    const float s = bng[o] * rsqrtf(bnv[o] + 1e-5f);
    float t = (acc[r] + convb[o] - bnm[o]) * s + bnb[o];
    t = fmaxf(t, 0.f);
#pragma unroll
    for (int m = 1; m < 16; m <<= 1) t += __shfl_xor(t, m, 32);  // sum over 16 spatial cols
    if (lm == 0) x_bf[(size_t)n * 256 + o] = (bf16)(t * (1.f / 16.f));
  }
}

// ---------------------------------------------------------------------------
// Flash-style multi-head attention, B=1, N=4096, heads=8, hd=32.
// Block = 8 waves = 8 query tiles of ONE head. Per 64-key chunk the block
// cooperatively stages K (row-major [j][d]) and V (transposed [d][j]) into
// LDS once, then every wave consumes the chunk: S = 2 WMMAs per 32-key
// subchunk (hd=32 is exactly one bf16 K-step), online softmax with per-row
// stats lane-local (D-layout rows == shfl half-groups), P transposed
// D->A layout through per-wave LDS (s_wait_dscnt orders it), O += P@V.
// K/V L2 traffic: 8x lower than wave-private streaming (~128MB total), and
// the V fragment becomes two contiguous ds_load_b128 thanks to the LDS
// transpose. Never materializes the 512MB score matrix.
// ---------------------------------------------------------------------------
__global__ void attention_kernel(const bf16* __restrict__ qkv, bf16* __restrict__ ao) {
  __shared__ __attribute__((aligned(16))) bf16 kbuf[64 * 32];        // [j][d]
  __shared__ __attribute__((aligned(16))) bf16 vtbuf[32 * 64];       // [d][j]
  __shared__ __attribute__((aligned(16))) bf16 pbuf[8][16 * 32];     // per-wave P

  const int tid  = threadIdx.x;       // 0..255
  const int wv   = tid >> 5;
  const int lane = tid & 31;
  const int half = lane >> 4;
  const int lm   = lane & 15;
  const int tq   = blockIdx.x * 8 + wv;
  const int h    = blockIdx.y;
  const int ROW  = 768;                       // qkv row: [q(256) | k(256) | v(256)]
  const float scale = 0.17677669529663687f;   // 32^-0.5

  const bf16* qrow = qkv + (size_t)(tq * 16 + lm) * ROW + h * 32;
  const v16bf qa = cat8(*(const v8bf*)(qrow + 16 * half),
                        *(const v8bf*)(qrow + 16 * half + 8));

  v8f o0 = v8f_zero(), o1 = v8f_zero();
  float mx[8], ls[8];
#pragma unroll
  for (int r = 0; r < 8; ++r) { mx[r] = -1e30f; ls[r] = 0.f; }

  bf16* pb = pbuf[wv];

  const int jr = tid >> 2;            // 0..63 : key row within chunk
  const int dc = (tid & 3) * 8;       // 0,8,16,24 : d-offset (8 bf16 = 16B)

  for (int j0 = 0; j0 < 4096; j0 += 64) {
    // ---- cooperative staging of the 64-key chunk ------------------------
    {
      const bf16* kr = qkv + (size_t)(j0 + jr) * ROW + 256 + h * 32 + dc;
      *(v8bf*)(kbuf + jr * 32 + dc) = *(const v8bf*)kr;
      const bf16* vr = qkv + (size_t)(j0 + jr) * ROW + 512 + h * 32 + dc;
      v8bf vv = *(const v8bf*)vr;
#pragma unroll
      for (int i = 0; i < 8; ++i) vtbuf[(dc + i) * 64 + jr] = vv[i];  // transpose
    }
    __syncthreads();

#pragma unroll
    for (int c = 0; c < 2; ++c) {
      const int jc = c * 32;

      // ---- scores: S[16q x 32j] as two 16x16 WMMA tiles (from LDS K) ----
      const bf16* k0r = kbuf + (jc + lm) * 32 + 16 * half;
      const bf16* k1r = kbuf + (jc + 16 + lm) * 32 + 16 * half;
      v16bf kb0 = cat8(*(const v8bf*)k0r, *(const v8bf*)(k0r + 8));
      v16bf kb1 = cat8(*(const v8bf*)k1r, *(const v8bf*)(k1r + 8));
      v8f s0 = WMMA_BF16(qa, kb0, v8f_zero());
      v8f s1 = WMMA_BF16(qa, kb1, v8f_zero());

      // ---- online softmax (per-row stats; rows live in 16-lane halves) --
#pragma unroll
      for (int r = 0; r < 8; ++r) {
        s0[r] *= scale; s1[r] *= scale;
        float v = fmaxf(s0[r], s1[r]);
#pragma unroll
        for (int m = 1; m < 16; m <<= 1) v = fmaxf(v, __shfl_xor(v, m, 32));
        const float mnew  = fmaxf(mx[r], v);
        const float alpha = __expf(mx[r] - mnew);
        mx[r] = mnew;
        const float p0 = __expf(s0[r] - mnew);
        const float p1 = __expf(s1[r] - mnew);
        s0[r] = p0; s1[r] = p1;
        float ps = p0 + p1;
#pragma unroll
        for (int m = 1; m < 16; m <<= 1) ps += __shfl_xor(ps, m, 32);
        ls[r] = ls[r] * alpha + ps;
        o0[r] *= alpha; o1[r] *= alpha;
      }

      // ---- P: D-layout -> A-layout via per-wave LDS bounce --------------
#pragma unroll
      for (int r = 0; r < 8; ++r) {
        const int row = r + 8 * half;
        pb[row * 32 + lm]      = (bf16)s0[r];
        pb[row * 32 + 16 + lm] = (bf16)s1[r];
      }
      asm volatile("s_wait_dscnt 0x0" ::: "memory");  // LDS stores visible before reads
      v16bf pa = cat8(*(const v8bf*)(pb + lm * 32 + 16 * half),
                      *(const v8bf*)(pb + lm * 32 + 16 * half + 8));

      // ---- O += P @ V : V fragment contiguous thanks to LDS transpose ---
      const bf16* v0r = vtbuf + lm * 64 + jc + 16 * half;
      const bf16* v1r = vtbuf + (16 + lm) * 64 + jc + 16 * half;
      v16bf vb0 = cat8(*(const v8bf*)v0r, *(const v8bf*)(v0r + 8));
      v16bf vb1 = cat8(*(const v8bf*)v1r, *(const v8bf*)(v1r + 8));
      o0 = WMMA_BF16(pa, vb0, o0);
      o1 = WMMA_BF16(pa, vb1, o1);
    }
    __syncthreads();   // chunk fully consumed before next staging overwrite
  }

  // ---- finalize: divide by row sum, store [N, HID] with col = h*32 + d ---
#pragma unroll
  for (int r = 0; r < 8; ++r) {
    const float  inv = 1.f / ls[r];
    const size_t q   = (size_t)tq * 16 + r + 8 * half;
    ao[q * 256 + h * 32 + lm]      = (bf16)(o0[r] * inv);
    ao[q * 256 + h * 32 + 16 + lm] = (bf16)(o1[r] * inv);
  }
}

// ------------------------------ small kernels ------------------------------

__global__ void cvt_bf16_kernel(const float* __restrict__ in, bf16* __restrict__ out, int n) {
  int i = blockIdx.x * blockDim.x + threadIdx.x;
  if (i < n) out[i] = (bf16)in[i];
}

__global__ void zero_f32_kernel(float* __restrict__ p, int n) {
  int i = blockIdx.x * blockDim.x + threadIdx.x;
  if (i < n) p[i] = 0.f;
}

__global__ void degree_kernel(const long long* __restrict__ edges, const float* __restrict__ w,
                              float* __restrict__ deg, int E) {
  int e = blockIdx.x * blockDim.x + threadIdx.x;
  if (e < E) {
    const int r = (int)edges[2 * e], c = (int)edges[2 * e + 1];
    const float wv = w[e];
    atomicAdd(&deg[r], wv);
    atomicAdd(&deg[c], wv);
  }
}

__global__ void dinv_kernel(float* __restrict__ deg, int n) {
  int i = blockIdx.x * blockDim.x + threadIdx.x;
  if (i < n) deg[i] = rsqrtf(deg[i] + 1e-6f);
}

// One block per undirected edge; thread f handles feature f for both directions.
__global__ void spmm_kernel(const long long* __restrict__ edges, const float* __restrict__ w,
                            const float* __restrict__ dinv,
                            const float* __restrict__ src, float* __restrict__ dst, int F) {
  const int e = blockIdx.x;
  const int f = threadIdx.x;
  const int r = (int)edges[2 * e], c = (int)edges[2 * e + 1];
  const float val = dinv[r] * w[e] * dinv[c];
  atomicAdd(&dst[(size_t)r * F + f], val * src[(size_t)c * F + f]);
  atomicAdd(&dst[(size_t)c * F + f], val * src[(size_t)r * F + f]);
}

__global__ void ln_relu_kernel(const float* __restrict__ in, const float* __restrict__ g,
                               const float* __restrict__ b,
                               float* __restrict__ outF, bf16* __restrict__ outB, int F) {
  __shared__ float red[256];
  const int row = blockIdx.x, t = threadIdx.x;
  const float v = in[(size_t)row * F + t];
  red[t] = v; __syncthreads();
  for (int s = F >> 1; s > 0; s >>= 1) { if (t < s) red[t] += red[t + s]; __syncthreads(); }
  const float mu = red[0] / F; __syncthreads();
  const float d = v - mu;
  red[t] = d * d; __syncthreads();
  for (int s = F >> 1; s > 0; s >>= 1) { if (t < s) red[t] += red[t + s]; __syncthreads(); }
  const float var = red[0] / F;
  float y = d * rsqrtf(var + 1e-5f) * g[t] + b[t];
  y = fmaxf(y, 0.f);
  outF[(size_t)row * F + t] = y;
  if (outB) outB[(size_t)row * F + t] = (bf16)y;
}

// ---------------------------------------------------------------------------

extern "C" void kernel_launch(void* const* d_in, const int* in_sizes, int n_in,
                              void* d_out, int out_size, void* d_ws, size_t ws_size,
                              hipStream_t stream) {
  (void)n_in; (void)out_size; (void)ws_size;

  const float*     node_feats = (const float*)d_in[0];
  const long long* edges      = (const long long*)d_in[1];   // int64 edge list
  const float*     edge_w     = (const float*)d_in[2];
  const float* conv_w = (const float*)d_in[3];
  const float* conv_b = (const float*)d_in[4];
  const float* bn_g   = (const float*)d_in[5];
  const float* bn_b   = (const float*)d_in[6];
  const float* bn_m   = (const float*)d_in[7];
  const float* bn_v   = (const float*)d_in[8];
  const float* qkv_w  = (const float*)d_in[9];
  const float* qkv_b  = (const float*)d_in[10];
  const float* proj_w = (const float*)d_in[11];
  const float* proj_b = (const float*)d_in[12];
  const float* g1_w   = (const float*)d_in[13];
  const float* g1_b   = (const float*)d_in[14];
  const float* ln1_g  = (const float*)d_in[15];
  const float* ln1_b  = (const float*)d_in[16];
  const float* g2_w   = (const float*)d_in[17];
  const float* g2_b   = (const float*)d_in[18];
  const float* ln2_g  = (const float*)d_in[19];
  const float* ln2_b  = (const float*)d_in[20];
  const float* sc_w   = (const float*)d_in[21];
  const float* sc_b   = (const float*)d_in[22];

  const int E  = in_sizes[2];     // 65536
  float* out   = (float*)d_out;   // [4096, 128]

  // ---- workspace carve-out (~33 MB) ----
  char*  ws  = (char*)d_ws;
  size_t off = 0;
  auto alloc = [&](size_t bytes) -> char* {
    char* p = ws + off;
    off += (bytes + 255) & ~(size_t)255;
    return p;
  };
  bf16*  convw_bf = (bf16*)alloc(256 * 256 * 2);
  bf16*  qkvw_bf  = (bf16*)alloc(768 * 256 * 2);
  bf16*  projw_bf = (bf16*)alloc(256 * 256 * 2);
  bf16*  g1w_bf   = (bf16*)alloc(256 * 256 * 2);
  bf16*  g2w_bf   = (bf16*)alloc(128 * 256 * 2);
  bf16*  scw_bf   = (bf16*)alloc(128 * 256 * 2);
  bf16*  x_bf     = (bf16*)alloc((size_t)4096 * 256 * 2);
  bf16*  qkv_bf   = (bf16*)alloc((size_t)4096 * 768 * 2);
  bf16*  attn_bf  = (bf16*)alloc((size_t)4096 * 256 * 2);
  float* x2       = (float*)alloc((size_t)4096 * 256 * 4);
  float* deg      = (float*)alloc(4096 * 4);
  float* spmm     = (float*)alloc((size_t)4096 * 256 * 4);   // reused for both SpMMs
  bf16*  spmm_bf  = (bf16*)alloc((size_t)4096 * 256 * 2);
  float* pre      = (float*)alloc((size_t)4096 * 256 * 4);   // reused pre-LN buffer
  float* g1f      = (float*)alloc((size_t)4096 * 256 * 4);
  bf16*  g1_bf    = (bf16*)alloc((size_t)4096 * 256 * 2);
  float* g2f      = (float*)alloc((size_t)4096 * 128 * 4);

  auto cvt = [&](const float* s, bf16* d, int n) {
    cvt_bf16_kernel<<<(n + 255) / 256, 256, 0, stream>>>(s, d, n);
  };

  // ---- weights -> bf16 (once per call) ----
  cvt(conv_w, convw_bf, 256 * 256);
  cvt(qkv_w,  qkvw_bf,  768 * 256);
  cvt(proj_w, projw_bf, 256 * 256);
  cvt(g1_w,   g1w_bf,   256 * 256);
  cvt(g2_w,   g2w_bf,   128 * 256);
  cvt(sc_w,   scw_bf,   128 * 256);

  // ---- 1) dim_reduce: conv1x1 + BN + ReLU + spatial mean -> x_bf [4096,256]
  dimreduce_kernel<<<4096, 512, 0, stream>>>(node_feats, convw_bf, conv_b,
                                             bn_g, bn_b, bn_m, bn_v, x_bf);

  // ---- 2) qkv projection -> qkv_bf [4096,768]
  gemm_bf16_kernel<<<dim3(256, 12), 128, 0, stream>>>(x_bf, qkvw_bf, qkv_b,
                                                      nullptr, qkv_bf, nullptr, 256, 768);

  // ---- 3) flash attention (LDS-staged K/V per 8-wave block) -> attn_bf
  attention_kernel<<<dim3(32, 8), 256, 0, stream>>>(qkv_bf, attn_bf);

  // ---- 4) output projection -> x2 f32 [4096,256]
  gemm_bf16_kernel<<<dim3(256, 4), 128, 0, stream>>>(attn_bf, projw_bf, proj_b,
                                                     x2, nullptr, nullptr, 256, 256);

  // ---- 5) normalized adjacency ----
  zero_f32_kernel<<<(4096 + 255) / 256, 256, 0, stream>>>(deg, 4096);
  degree_kernel<<<(E + 255) / 256, 256, 0, stream>>>(edges, edge_w, deg, E);
  dinv_kernel<<<(4096 + 255) / 256, 256, 0, stream>>>(deg, 4096);   // deg := (deg+eps)^-1/2

  // ---- 6) GCN layer 1 ----
  zero_f32_kernel<<<(4096 * 256 + 255) / 256, 256, 0, stream>>>(spmm, 4096 * 256);
  spmm_kernel<<<E, 256, 0, stream>>>(edges, edge_w, deg, x2, spmm, 256);
  cvt(spmm, spmm_bf, 4096 * 256);
  gemm_bf16_kernel<<<dim3(256, 4), 128, 0, stream>>>(spmm_bf, g1w_bf, g1_b,
                                                     pre, nullptr, nullptr, 256, 256);
  ln_relu_kernel<<<4096, 256, 0, stream>>>(pre, ln1_g, ln1_b, g1f, g1_bf, 256);

  // ---- 7) GCN layer 2 ----
  zero_f32_kernel<<<(4096 * 256 + 255) / 256, 256, 0, stream>>>(spmm, 4096 * 256);
  spmm_kernel<<<E, 256, 0, stream>>>(edges, edge_w, deg, g1f, spmm, 256);
  cvt(spmm, spmm_bf, 4096 * 256);
  gemm_bf16_kernel<<<dim3(256, 2), 128, 0, stream>>>(spmm_bf, g2w_bf, g2_b,
                                                     pre, nullptr, nullptr, 256, 128);
  ln_relu_kernel<<<4096, 128, 0, stream>>>(pre, ln2_g, ln2_b, g2f, nullptr, 128);

  // ---- 8) shortcut GEMM fused with residual: out = g2 + g1 @ sc_w^T + sc_b
  gemm_bf16_kernel<<<dim3(256, 2), 128, 0, stream>>>(g1_bf, scw_bf, sc_b,
                                                     out, nullptr, g2f, 256, 128);
}
